// VectorQuantizer_49804440764749
// MI455X (gfx1250) — matrix operand order repaired and checked by reading, not compiled, not run on hardware
//
#include <hip/hip_runtime.h>
#include <hip/hip_bf16.h>
#include <stdint.h>

// Problem dimensions (from reference: x (32,32,32,256), codebook (1024,256))
#define NS 32768       // flattened samples
#define DD 256         // feature dim
#define KC 1024        // codebook entries
#define MT 32          // sample rows per workgroup in the main kernel
#define XS_PITCH 260   // LDS pitch for x tile (f32): 1040B row stride =
                       // 16B-aligned for async b128, banks rotate 4/row
#define DIST_PITCH 1028// LDS pitch for distance tile (f32), bank rotate 4/row

#define COMMIT_W 0.25f
#define ENT_W    0.1f
#define INV_T    100.0f   // 1 / ENT_T
#define EPSF     1e-5f

typedef __bf16 v16bf __attribute__((ext_vector_type(16)));
typedef float  v8f   __attribute__((ext_vector_type(8)));

__device__ __forceinline__ float wred_add(float v) {
#pragma unroll
  for (int off = 16; off >= 1; off >>= 1) v += __shfl_xor(v, off, 32);
  return v;
}

// ---------------------------------------------------------------------------
// K0: zero the global accumulators (must run every launch: ws is not
// re-poisoned between replays and we accumulate with atomics).
// ---------------------------------------------------------------------------
extern "C" __global__ void vq_init(float* __restrict__ avgp,
                                   float* __restrict__ scal) {
  int t = blockIdx.x * blockDim.x + threadIdx.x;
  if (t < KC) avgp[t] = 0.0f;
  if (t < 2)  scal[t] = 0.0f;
}

// ---------------------------------------------------------------------------
// K1: codebook prep — exact f32 row norms + bf16 hi/lo split (ch + cl == c
// to ~16 mantissa bits). One wave per codebook row.
// ---------------------------------------------------------------------------
extern "C" __global__ __launch_bounds__(256)
void vq_prep(const float* __restrict__ cb,
             __bf16* __restrict__ ch, __bf16* __restrict__ cl,
             float* __restrict__ cnorm) {
  const int wave = threadIdx.x >> 5, lane = threadIdx.x & 31;
  const int r = blockIdx.x * 8 + wave;
  float s = 0.0f;
#pragma unroll
  for (int j = 0; j < DD / 32; ++j) {
    const int d = lane + 32 * j;
    const float c = cb[r * DD + d];
    const __bf16 h = (__bf16)c;
    ch[r * DD + d] = h;
    cl[r * DD + d] = (__bf16)(c - (float)h);
    s += c * c;
  }
  s = wred_add(s);
  if (lane == 0) cnorm[r] = s;
}

// ---------------------------------------------------------------------------
// K2: main fused kernel.
//   Per block: 32 samples x 1024 codes, distance tile lives in LDS only.
//   GEMM: 3-pass bf16 WMMA (xh*ch + xh*cl + xl*ch), f32 accumulate.
//   Wave tiling: each wave covers BOTH 16-row M-tiles x 8 strided N-tiles,
//   so every B fragment read from L2 feeds 6 WMMAs (codebook read once/block).
// ---------------------------------------------------------------------------
extern "C" __global__ __launch_bounds__(256)
void vq_main(const float* __restrict__ x, const float* __restrict__ cb,
             const __bf16* __restrict__ ch, const __bf16* __restrict__ cl,
             const float* __restrict__ cnorm,
             float* __restrict__ avgp, float* __restrict__ scal,
             float* __restrict__ outq, float* __restrict__ outidx) {
  extern __shared__ char smem[];
  float* xs   = (float*)smem;                              // MT*XS_PITCH f32
  float* dist = (float*)(smem + MT * XS_PITCH * 4);        // MT*DIST_PITCH f32
  __shared__ float xnr[MT];
  __shared__ float sinv[MT];
  __shared__ int   irow[MT];

  const int tid  = threadIdx.x;
  const int wave = tid >> 5;
  const int lane = tid & 31;
  const int row0 = blockIdx.x * MT;

  // ---- stage x tile: CDNA5 async global->LDS (no VGPR round-trip) ----
  // 2048 16-byte units (32 rows x 64 units); 8 units per thread, coalesced.
  {
    const uint32_t xs_lds = (uint32_t)(uintptr_t)xs;  // LDS aperture: low 32b
#pragma unroll
    for (int p = 0; p < 8; ++p) {
      const int u = p * 256 + tid;
      const int r = u >> 6;          // row in tile
      const int c = u & 63;          // 16B chunk within row
      const uint64_t gaddr =
          (uint64_t)(uintptr_t)(x + (size_t)(row0 + r) * DD + c * 4);
      const uint32_t laddr = xs_lds + (uint32_t)(r * XS_PITCH * 4 + c * 16);
      asm volatile("global_load_async_to_lds_b128 %0, %1, off"
                   :: "v"(laddr), "v"(gaddr) : "memory");
    }
    asm volatile("s_wait_asynccnt 0x0" ::: "memory");
  }
  __syncthreads();

  // ---- exact f32 row norms (4 rows per wave) ----
#pragma unroll
  for (int q = 0; q < MT / 8; ++q) {
    const int r = wave * (MT / 8) + q;
    float s = 0.0f;
#pragma unroll
    for (int j = 0; j < DD / 32; ++j) {
      const float f = xs[r * XS_PITCH + lane + 32 * j];
      s += f * f;
    }
    s = wred_add(s);
    if (lane == 0) xnr[r] = s;
  }
  __syncthreads();

  // ---- WMMA GEMM: dot[m][n] = sum_d x[m][d]*c[n][d] ----
  const int grp = lane >> 4;   // lane group 0/1
  const int lm  = lane & 15;

  v8f acc[16] = {};            // [mi*8 + t]
  for (int dc = 0; dc < 8; ++dc) {          // feature chunks of 32
    const int kb = dc * 32;
    // Build A fragments for both M-tiles (16x32 bf16, contiguous k = 16*grp + i, same order as B).
    v16bf ah[2], al[2];
#pragma unroll
    for (int mi = 0; mi < 2; ++mi) {
#pragma unroll
      for (int i = 0; i < 16; ++i) {
        const int d = kb + grp * 16 + i;
        const float f = xs[(mi * 16 + lm) * XS_PITCH + d];
        const __bf16 h = (__bf16)f;
        ah[mi][i] = h;
        al[mi][i] = (__bf16)(f - (float)h);
      }
    }
#pragma unroll
    for (int t = 0; t < 8; ++t) {
      const int n = (wave + 8 * t) * 16 + lm; // this lane's B column (code id)
      // B fragment (32x16 bf16): lane group selects K half, 16 contiguous d's.
      const v16bf bh = *(const v16bf*)(ch + (size_t)n * DD + kb + grp * 16);
      const v16bf bl = *(const v16bf*)(cl + (size_t)n * DD + kb + grp * 16);
#pragma unroll
      for (int mi = 0; mi < 2; ++mi) {
        acc[mi * 8 + t] = __builtin_amdgcn_wmma_f32_16x16x32_bf16(
            false, ah[mi], false, bh, (short)0, acc[mi * 8 + t], false, false);
        acc[mi * 8 + t] = __builtin_amdgcn_wmma_f32_16x16x32_bf16(
            false, ah[mi], false, bl, (short)0, acc[mi * 8 + t], false, false);
        acc[mi * 8 + t] = __builtin_amdgcn_wmma_f32_16x16x32_bf16(
            false, al[mi], false, bh, (short)0, acc[mi * 8 + t], false, false);
      }
    }
  }

  // ---- dot -> distance, spill tile to LDS ----
#pragma unroll
  for (int t = 0; t < 8; ++t) {
    const int n  = (wave + 8 * t) * 16 + lm;
    const float cn = cnorm[n];
#pragma unroll
    for (int mi = 0; mi < 2; ++mi) {
#pragma unroll
      for (int r = 0; r < 8; ++r) {
        const int m = mi * 16 + 8 * grp + r;  // C layout: lanes16-31 hold M+8
        const float d2 = xnr[m] + cn - 2.0f * acc[mi * 8 + t][r];
        dist[m * DIST_PITCH + n] = sqrtf(fmaxf(d2, 0.0f));
      }
    }
  }
  __syncthreads();

  // ---- per-row argmin + online softmax stats (4 rows per wave) ----
  float went = 0.0f;   // sum over rows of sum_k p*log p (lane 0 only)
#pragma unroll
  for (int q = 0; q < MT / 8; ++q) {
    const int m = wave * (MT / 8) + q;
    float bd = 3.4e38f; int bk = 0;
    for (int j = 0; j < KC / 32; ++j) {
      const int k = lane + 32 * j;
      const float d = dist[m * DIST_PITCH + k];
      if (d < bd) { bd = d; bk = k; }       // k ascending -> first occurrence
    }
#pragma unroll
    for (int off = 16; off >= 1; off >>= 1) {
      const float od = __shfl_xor(bd, off, 32);
      const int   ok = __shfl_xor(bk, off, 32);
      if (od < bd || (od == bd && ok < bk)) { bd = od; bk = ok; }
    }
    const float mx = -INV_T * bd;           // row max logit (logit = -d/T)
    float s = 0.0f, tt = 0.0f;
    for (int j = 0; j < KC / 32; ++j) {
      const int k = lane + 32 * j;
      const float lg = -INV_T * dist[m * DIST_PITCH + k] - mx;   // <= 0
      const float e  = __expf(lg);
      s += e; tt += e * lg;
      dist[m * DIST_PITCH + k] = e;         // overwrite with unnormalized prob
    }
    s  = wred_add(s);
    tt = wred_add(tt);
    if (lane == 0) {
      sinv[m] = 1.0f / s;
      irow[m] = bk;
      outidx[row0 + m] = (float)bk;
      went += tt / s - __logf(s);           // sum_k p*log p (EPS shift cancels)
    }
  }
  if (lane == 0) atomicAdd(&scal[1], went);
  __syncthreads();

  // ---- avg_probs partials: column-parallel, no LDS races ----
#pragma unroll
  for (int i = 0; i < KC / 256; ++i) {
    const int k = tid + 256 * i;
    float a = 0.0f;
#pragma unroll
    for (int m = 0; m < MT; ++m) a += dist[m * DIST_PITCH + k] * sinv[m];
    atomicAdd(&avgp[k], a);                 // 1 atomic per (k, block)
  }

  // ---- quantized gather (exact f32 codebook) + squared error ----
  float err = 0.0f;
#pragma unroll
  for (int r = 0; r < MT; ++r) {
    const float qv = cb[(size_t)irow[r] * DD + tid];
    outq[(size_t)(row0 + r) * DD + tid] = qv;
    const float df = xs[r * XS_PITCH + tid] - qv;
    err += df * df;
  }
  err = wred_add(err);
  if (lane == 0) atomicAdd(&scal[0], err);
}

// ---------------------------------------------------------------------------
// K3: fold accumulators into the 4 loss scalars.
// ---------------------------------------------------------------------------
extern "C" __global__ __launch_bounds__(256)
void vq_final(const float* __restrict__ avgp, const float* __restrict__ scal,
              float* __restrict__ outs) {
  __shared__ float red[8];
  const int tid = threadIdx.x, lane = tid & 31, wave = tid >> 5;
  float a = 0.0f;
#pragma unroll
  for (int i = 0; i < KC / 256; ++i) {
    const int k = tid + 256 * i;
    const float p = avgp[k] * (1.0f / (float)NS);
    a += p * __logf(p + EPSF);
  }
  a = wred_add(a);
  if (lane == 0) red[wave] = a;
  __syncthreads();
  if (tid == 0) {
    float tot = 0.0f;
#pragma unroll
    for (int w = 0; w < 8; ++w) tot += red[w];
    const float avg_ent  = -tot;
    const float samp_ent = -scal[1] * (1.0f / (float)NS);
    const float ent_loss = (samp_ent - avg_ent) * ENT_W;
    const float mse_half = 0.5f * scal[0] / (float)((size_t)NS * DD);
    const float commit   = mse_half * COMMIT_W;
    const float cb_loss  = mse_half;
    outs[0] = cb_loss + commit + ent_loss;  // vq_loss
    outs[1] = commit;
    outs[2] = cb_loss;
    outs[3] = ent_loss;
  }
}

// ---------------------------------------------------------------------------
extern "C" void kernel_launch(void* const* d_in, const int* in_sizes, int n_in,
                              void* d_out, int out_size, void* d_ws, size_t ws_size,
                              hipStream_t stream) {
  (void)in_sizes; (void)n_in; (void)out_size; (void)ws_size;
  const float* x  = (const float*)d_in[0];   // (32,32,32,256) f32
  const float* cb = (const float*)d_in[1];   // (1024,256)     f32

  // Workspace: ch (512KB) | cl (512KB) | cnorm (4KB) | avg_probs (4KB) | scal(2)
  char*   ws    = (char*)d_ws;
  __bf16* ch    = (__bf16*)ws;
  __bf16* cl    = (__bf16*)(ws + (size_t)KC * DD * 2);
  float*  cnorm = (float*)(ws + (size_t)KC * DD * 4);
  float*  avgp  = cnorm + KC;
  float*  scal  = avgp + KC;

  // d_out: quantized (NS*DD) | vq,commit,codebook,entropy (4) | indices (NS)
  float* outq   = (float*)d_out;
  float* outs   = outq + (size_t)NS * DD;
  float* outidx = outs + 4;

  vq_init<<<4, 256, 0, stream>>>(avgp, scal);
  vq_prep<<<KC / 8, 256, 0, stream>>>(cb, ch, cl, cnorm);

  const size_t smem = (size_t)MT * XS_PITCH * 4      // x tile   (33,280 B)
                    + (size_t)MT * DIST_PITCH * 4;   // distances(131,584 B)
  vq_main<<<NS / MT, 256, smem, stream>>>(x, cb, ch, cl, cnorm, avgp, scal,
                                          outq, outidx);
  vq_final<<<1, 256, 0, stream>>>(avgp, scal, outs);
}
